// VENet_2027224564042
// MI455X (gfx1250) — matrix operand order repaired and checked by reading, not compiled, not run on hardware
//
#include <hip/hip_runtime.h>
#include <math.h>

// ---------------------------------------------------------------------------
// Model dims
// ---------------------------------------------------------------------------
#define T_DIM 4096
#define F_DIM 4096
#define H1    250
#define H1P   256      // padded per-direction hidden
#define HCAT  512      // padded concat(fwd,bwd) = 500 -> 512
#define MLP1  50
#define GH    2048
#define G3    6144     // 3*GH
#define EMB   100
#define NC    6
#define GRU_NWG 16

#define BPAD  40       // LDS B-tile row pitch in f16 (80B: 16B-aligned, conflict-free)

typedef __attribute__((ext_vector_type(16))) _Float16 v16h;
typedef __attribute__((ext_vector_type(8)))  _Float16 h8;
typedef __attribute__((ext_vector_type(8)))  float    v8f;

// ---------------------------------------------------------------------------
// Conversions / packing (f32 -> f16, with zero padding for tile-friendly dims)
// ---------------------------------------------------------------------------
__global__ void k_relu_f16(const float* __restrict__ x, _Float16* __restrict__ o, int n) {
    int i = blockIdx.x * blockDim.x + threadIdx.x;
    if (i < n) o[i] = (_Float16)fmaxf(x[i], 0.f);
}

__global__ void k_cvt_f16(const float* __restrict__ x, _Float16* __restrict__ o, int n) {
    int i = blockIdx.x * blockDim.x + threadIdx.x;
    if (i < n) o[i] = (_Float16)x[i];
}

// ind_w0 (2,250,4096) -> [512][4096] f16, rows d*256+h, zero-padded
__global__ void k_pack_w0(const float* __restrict__ w, _Float16* __restrict__ o) {
    int i = blockIdx.x * blockDim.x + threadIdx.x;     // over 512*4096
    int k = i & (F_DIM - 1);
    int r = i >> 12;
    int d = r >> 8, h = r & 255;
    float v = (h < H1) ? w[((size_t)(d * H1 + h)) * F_DIM + k] : 0.f;
    o[i] = (_Float16)v;
}

// ind_w1 (2,250,500) -> [512][512] f16
__global__ void k_pack_w1(const float* __restrict__ w, _Float16* __restrict__ o) {
    int i = blockIdx.x * blockDim.x + threadIdx.x;     // over 512*512
    int c = i & (HCAT - 1);
    int r = i >> 9;
    int d = r >> 8, h = r & 255;
    float v = (h < H1 && c < 2 * H1) ? w[((size_t)(d * H1 + h)) * (2 * H1) + c] : 0.f;
    o[i] = (_Float16)v;
}

// ---------------------------------------------------------------------------
// WMMA GEMM:  C[M x N](f32) = A[M x K](f16, row-major) * W[N x K](f16)^T
//
// WG = 8 waves covering 128 rows x 64 cols. All waves share the same B strip,
// so the B tile (64 x 32 f16 per K-step) is staged once per WG into LDS via
// GLOBAL_LOAD_ASYNC_TO_LDS_B128 (ASYNCcnt), double-buffered; WMMA B operands
// come from LDS (ds_load_b128). A operands stream directly from global
// (unique rows per wave). Fragment layouts per CDNA5 ISA 7.12.2 (wave32).
// ---------------------------------------------------------------------------
__global__ void __launch_bounds__(256)
k_wmma_gemm(const _Float16* __restrict__ A, int lda,
            const _Float16* __restrict__ W, int ldw,
            float* __restrict__ C, int ldc, int K) {
    __shared__ _Float16 bsm[2][64][BPAD];

    const int lane = threadIdx.x & 31;
    const int wid  = threadIdx.x >> 5;
    const int m0   = (blockIdx.y * 8 + wid) * 16;
    const int n0   = blockIdx.x * 64;
    const int r    = lane & 15;
    const int sel  = lane >> 4;          // K-half select

    // async staging assignment: thread -> one 16B chunk of the 64x32 B tile
    const int srow = threadIdx.x >> 2;       // 0..63
    const int schk = threadIdx.x & 3;        // 0..3 (16B chunks of a 64B row)
    const _Float16* gsrc0 = W + (size_t)(n0 + srow) * (size_t)ldw + schk * 8;

    v8f acc0 = {}, acc1 = {}, acc2 = {}, acc3 = {};

    const _Float16* arow = A + (size_t)(m0 + r) * (size_t)lda;
    const int nsteps = K >> 5;

    // prologue: stage K-step 0 into buffer 0
    {
        unsigned ldst = (unsigned)(uintptr_t)&bsm[0][srow][schk * 8];
        unsigned long long ga = (unsigned long long)(uintptr_t)gsrc0;
        asm volatile("global_load_async_to_lds_b128 %0, %1, off"
                     :: "v"(ldst), "v"(ga) : "memory");
    }

    for (int ki = 0; ki < nsteps; ++ki) {
        const int buf = ki & 1;
        // my async write (and everyone else's, after the barrier) is complete
        asm volatile("s_wait_asynccnt 0x0" ::: "memory");
        __syncthreads();

        if (ki + 1 < nsteps) {   // stage next K-step into the other buffer
            unsigned ldst = (unsigned)(uintptr_t)&bsm[buf ^ 1][srow][schk * 8];
            unsigned long long ga =
                (unsigned long long)(uintptr_t)(gsrc0 + (size_t)(ki + 1) * 32);
            asm volatile("global_load_async_to_lds_b128 %0, %1, off"
                         :: "v"(ldst), "v"(ga) : "memory");
        }

        const int kk = ki * 32;
        union Frag { v16h v; struct { h8 lo, hi; } s; } a, b;
        a.s.lo = *(const h8*)(arow + kk +      sel * 8);   // K = sel*8 .. +7
        a.s.hi = *(const h8*)(arow + kk + 16 + sel * 8);   // K = 16+sel*8 .. +7

        const _Float16* bp0 = &bsm[buf][ 0 + r][sel * 8];
        const _Float16* bp1 = &bsm[buf][16 + r][sel * 8];
        const _Float16* bp2 = &bsm[buf][32 + r][sel * 8];
        const _Float16* bp3 = &bsm[buf][48 + r][sel * 8];

        b.s.lo = *(const h8*)bp0;  b.s.hi = *(const h8*)(bp0 + 16);
        acc0 = __builtin_amdgcn_wmma_f32_16x16x32_f16(false, a.v, false, b.v,
                                                      (short)0, acc0, false, false);
        b.s.lo = *(const h8*)bp1;  b.s.hi = *(const h8*)(bp1 + 16);
        acc1 = __builtin_amdgcn_wmma_f32_16x16x32_f16(false, a.v, false, b.v,
                                                      (short)0, acc1, false, false);
        b.s.lo = *(const h8*)bp2;  b.s.hi = *(const h8*)(bp2 + 16);
        acc2 = __builtin_amdgcn_wmma_f32_16x16x32_f16(false, a.v, false, b.v,
                                                      (short)0, acc2, false, false);
        b.s.lo = *(const h8*)bp3;  b.s.hi = *(const h8*)(bp3 + 16);
        acc3 = __builtin_amdgcn_wmma_f32_16x16x32_f16(false, a.v, false, b.v,
                                                      (short)0, acc3, false, false);
    }

    // D: VGPR v holds M = m0 + v + 8*sel, N = n0 + (lane&15) (+16 per tile)
    float* crow = C + (size_t)(m0 + sel * 8) * (size_t)ldc + n0 + r;
#pragma unroll
    for (int v = 0; v < 8; ++v) {
        crow[(size_t)v * ldc +  0] = acc0[v];
        crow[(size_t)v * ldc + 16] = acc1[v];
        crow[(size_t)v * ldc + 32] = acc2[v];
        crow[(size_t)v * ldc + 48] = acc3[v];
    }
}

// ---------------------------------------------------------------------------
// IndRNN diagonal scan: 512 independent serial channels (1 thread each).
// xw is [T][512] f32 (cols 0..255 = dir0 padded, 256..511 = dir1 padded).
// Output col c in 0..499 = concat(fwd 250, bwd 250); pad cols zeroed.
// ---------------------------------------------------------------------------
__global__ void __launch_bounds__(512)
k_indrnn_scan(const float* __restrict__ xw,
              const float* __restrict__ u, const float* __restrict__ b,
              _Float16* __restrict__ oh16, float* __restrict__ of32) {
    int c = threadIdx.x;
    if (c >= 2 * H1) {                      // zero the pad columns
        for (int t = 0; t < T_DIM; ++t) {
            if (oh16) oh16[(size_t)t * HCAT + c] = (_Float16)0.f;
            if (of32) of32[(size_t)t * HCAT + c] = 0.f;
        }
        return;
    }
    int d  = (c < H1) ? 0 : 1;
    int ch = (d == 0) ? c : c - H1;
    int incol = d * H1P + ch;
    float uu = u[d * H1 + ch];
    float bb = b[d * H1 + ch];
    float h = 0.f;
    if (d == 0) {
        for (int t = 0; t < T_DIM; ++t) {
            h = fmaxf(xw[(size_t)t * HCAT + incol] + bb + uu * h, 0.f);
            if (oh16) oh16[(size_t)t * HCAT + c] = (_Float16)h;
            if (of32) of32[(size_t)t * HCAT + c] = h;
        }
    } else {
        for (int t = T_DIM - 1; t >= 0; --t) {
            h = fmaxf(xw[(size_t)t * HCAT + incol] + bb + uu * h, 0.f);
            if (oh16) oh16[(size_t)t * HCAT + c] = (_Float16)h;
            if (of32) of32[(size_t)t * HCAT + c] = h;
        }
    }
}

// ---------------------------------------------------------------------------
// MLP head -> pooled[t] (sigmoid). One block (64 threads) per timestep.
// ---------------------------------------------------------------------------
__global__ void __launch_bounds__(64)
k_mlp(const float* __restrict__ h2,
      const float* __restrict__ w1, const float* __restrict__ b1,
      const float* __restrict__ w2, const float* __restrict__ b2,
      float* __restrict__ pooled) {
    __shared__ float sm[MLP1];
    int t = blockIdx.x, i = threadIdx.x;
    if (i < MLP1) {
        const float* hp = h2 + (size_t)t * HCAT;
        const float* wr = w1 + (size_t)i * (2 * H1);
        float s = b1[i];
        for (int c = 0; c < 2 * H1; ++c) s += hp[c] * wr[c];
        sm[i] = fmaxf(s, 0.f);
    }
    __syncthreads();
    if (i == 0) {
        float s = b2[0];
        for (int k = 0; k < MLP1; ++k) s += sm[k] * w2[k];
        pooled[t] = 1.f / (1.f + expf(-s));
    }
}

// mask = any(pooled>0.5) ? sel : one-hot(0).  Single 1024-thread block.
__global__ void __launch_bounds__(1024)
k_mask(const float* __restrict__ pooled, float* __restrict__ mask) {
    __shared__ int any_sel;
    if (threadIdx.x == 0) any_sel = 0;
    __syncthreads();
    int sel[T_DIM / 1024];
    for (int j = 0; j < T_DIM / 1024; ++j) {
        int i = threadIdx.x + j * 1024;
        sel[j] = pooled[i] > 0.5f;
        if (sel[j]) any_sel = 1;
    }
    __syncthreads();
    int a = any_sel;
    for (int j = 0; j < T_DIM / 1024; ++j) {
        int i = threadIdx.x + j * 1024;
        mask[i] = a ? (float)sel[j] : ((i == 0) ? 1.f : 0.f);
    }
}

// ---------------------------------------------------------------------------
// GRU: grid-persistent scan. 16 WGs x 256 threads; 2 threads per hidden j
// (K split in half, combined via shfl_xor). whh kept in f16 (25 MB -> L2
// resident). Ping-pong h buffers; one device-scope barrier per step.
// ---------------------------------------------------------------------------
__global__ void __launch_bounds__(256)
k_gru(const float* __restrict__ gx, const float* __restrict__ bih,
      const _Float16* __restrict__ whh, const float* __restrict__ bhh,
      const float* __restrict__ mask, float* __restrict__ hbuf,
      unsigned* __restrict__ counter) {
    __shared__ float sh[GH];
    const int gtid = blockIdx.x * 256 + threadIdx.x;   // 0..4095
    const int j    = gtid >> 1;
    const int half = gtid & 1;
    const int kb   = half * (GH / 2);

    const _Float16* wr = whh + (size_t)j            * GH + kb;
    const _Float16* wz = whh + (size_t)(GH + j)     * GH + kb;
    const _Float16* wn = whh + (size_t)(2 * GH + j) * GH + kb;
    const float br = bhh[j], bz = bhh[GH + j], bn = bhh[2 * GH + j];
    const float cr = bih[j], cz = bih[GH + j], cn = bih[2 * GH + j];

    for (int t = 0; t < T_DIM; ++t) {
        const float* hb = hbuf + (t & 1) * GH;
        for (int i = threadIdx.x; i < GH; i += 256) sh[i] = hb[i];
        __syncthreads();

        if (t + 1 < T_DIM) {  // prefetch next step's gx row (global_prefetch_b8)
            __builtin_prefetch(gx + (size_t)(t + 1) * G3 + j, 0, 0);
        }

        float sr = 0.f, sz = 0.f, sn = 0.f;
        for (int k = 0; k < GH / 2; k += 8) {
            h8 r8 = *(const h8*)(wr + k);
            h8 z8 = *(const h8*)(wz + k);
            h8 n8 = *(const h8*)(wn + k);
#pragma unroll
            for (int l = 0; l < 8; ++l) {
                float hv = sh[kb + k + l];
                sr += (float)r8[l] * hv;
                sz += (float)z8[l] * hv;
                sn += (float)n8[l] * hv;
            }
        }
        sr += __shfl_xor(sr, 1, 32);
        sz += __shfl_xor(sz, 1, 32);
        sn += __shfl_xor(sn, 1, 32);

        const float* gxt = gx + (size_t)t * G3;
        float r = 1.f / (1.f + expf(-(gxt[j]          + cr + sr + br)));
        float z = 1.f / (1.f + expf(-(gxt[GH + j]     + cz + sz + bz)));
        float n = tanhf(        gxt[2 * GH + j] + cn + r * (sn + bn));
        float hp = sh[j];
        float hnew = (1.f - z) * n + z * hp;
        float hout = (mask[t] > 0.f) ? hnew : hp;
        if (half == 0) hbuf[((t + 1) & 1) * GH + j] = hout;

        // device-scope barrier (monotonic counter; zeroed per launch)
        __syncthreads();
        if (threadIdx.x == 0) {
            __threadfence();
            __hip_atomic_fetch_add(counter, 1u, __ATOMIC_ACQ_REL, __HIP_MEMORY_SCOPE_AGENT);
            unsigned target = (unsigned)(t + 1) * GRU_NWG;
            while (__hip_atomic_load(counter, __ATOMIC_ACQUIRE, __HIP_MEMORY_SCOPE_AGENT) < target) {
                __builtin_amdgcn_s_sleep(1);
            }
        }
        __syncthreads();
    }
}

// fc1 -> relu -> l2norm -> fc2. Single block.
__global__ void __launch_bounds__(128)
k_final(const float* __restrict__ hT,
        const float* __restrict__ fc1w, const float* __restrict__ fc1b,
        const float* __restrict__ fc2w, const float* __restrict__ fc2b,
        float* __restrict__ out) {
    __shared__ float se[EMB];
    __shared__ float snorm;
    int i = threadIdx.x;
    if (i < EMB) {
        const float* wr = fc1w + (size_t)i * GH;
        float s = fc1b[i];
        for (int k = 0; k < GH; ++k) s += hT[k] * wr[k];
        se[i] = fmaxf(s, 0.f);
    }
    __syncthreads();
    if (i == 0) {
        float s = 0.f;
        for (int k = 0; k < EMB; ++k) s += se[k] * se[k];
        snorm = sqrtf(s);
    }
    __syncthreads();
    if (i < NC) {
        const float* wr = fc2w + (size_t)i * EMB;
        float inv = 1.f / snorm;
        float s = fc2b[i];
        for (int k = 0; k < EMB; ++k) s += se[k] * inv * wr[k];
        out[i] = s;
    }
}

__global__ void k_init(unsigned* __restrict__ counter, float* __restrict__ hbuf) {
    int i = blockIdx.x * blockDim.x + threadIdx.x;
    if (i == 0) *counter = 0u;
    if (i < 2 * GH) hbuf[i] = 0.f;
}

// ---------------------------------------------------------------------------
// Orchestration
// ---------------------------------------------------------------------------
extern "C" void kernel_launch(void* const* d_in, const int* in_sizes, int n_in,
                              void* d_out, int out_size, void* d_ws, size_t ws_size,
                              hipStream_t stream) {
    (void)in_sizes; (void)n_in; (void)out_size; (void)ws_size;

    const float* app_x = (const float*)d_in[0];
    const float* w0    = (const float*)d_in[1];
    const float* u0    = (const float*)d_in[2];
    const float* b0    = (const float*)d_in[3];
    const float* w1    = (const float*)d_in[4];
    const float* u1    = (const float*)d_in[5];
    const float* b1    = (const float*)d_in[6];
    const float* mlp1w = (const float*)d_in[7];
    const float* mlp1b = (const float*)d_in[8];
    const float* mlp2w = (const float*)d_in[9];
    const float* mlp2b = (const float*)d_in[10];
    const float* wih   = (const float*)d_in[11];
    const float* whh   = (const float*)d_in[12];
    const float* bih   = (const float*)d_in[13];
    const float* bhh   = (const float*)d_in[14];
    const float* fc1w  = (const float*)d_in[15];
    const float* fc1b  = (const float*)d_in[16];
    const float* fc2w  = (const float*)d_in[17];
    const float* fc2b  = (const float*)d_in[18];
    float* out = (float*)d_out;

    char* ws = (char*)d_ws;
    size_t off = 0;
    auto walloc = [&](size_t bytes) -> void* {
        void* p = ws + off;
        off = (off + bytes + 255) & ~(size_t)255;
        return p;
    };

    unsigned*  counter = (unsigned*)walloc(256);
    float*     hbuf    = (float*)walloc((size_t)2 * GH * 4);
    _Float16*  xh      = (_Float16*)walloc((size_t)T_DIM * F_DIM * 2);
    _Float16*  w0p     = (_Float16*)walloc((size_t)HCAT * F_DIM * 2);
    _Float16*  w1p     = (_Float16*)walloc((size_t)HCAT * HCAT * 2);
    _Float16*  wihh    = (_Float16*)walloc((size_t)G3 * F_DIM * 2);
    _Float16*  whhh    = (_Float16*)walloc((size_t)G3 * GH * 2);
    float*     xw      = (float*)walloc((size_t)T_DIM * HCAT * 4);   // xw0 then xw1
    _Float16*  h1h     = (_Float16*)walloc((size_t)T_DIM * HCAT * 2);
    float*     h2f     = (float*)walloc((size_t)T_DIM * HCAT * 4);
    float*     gx      = (float*)walloc((size_t)T_DIM * G3 * 4);

    float* pooled = out + NC;
    float* maskp  = out + NC + T_DIM;

    k_init<<<dim3(4), dim3(1024), 0, stream>>>(counter, hbuf);
    k_relu_f16<<<dim3((T_DIM * F_DIM) / 1024), 1024, 0, stream>>>(app_x, xh, T_DIM * F_DIM);
    k_pack_w0<<<dim3((HCAT * F_DIM) / 256), 256, 0, stream>>>(w0, w0p);
    k_pack_w1<<<dim3((HCAT * HCAT) / 256), 256, 0, stream>>>(w1, w1p);
    k_cvt_f16<<<dim3((G3 * F_DIM) / 1024), 1024, 0, stream>>>(wih, wihh, G3 * F_DIM);
    k_cvt_f16<<<dim3((G3 * GH) / 1024), 1024, 0, stream>>>(whh, whhh, G3 * GH);

    // xw0 = relu(x) @ w0pad^T : [4096 x 4096] * [512 x 4096]^T
    k_wmma_gemm<<<dim3(HCAT / 64, T_DIM / 128), 256, 0, stream>>>(xh, F_DIM, w0p, F_DIM,
                                                                  xw, HCAT, F_DIM);
    k_indrnn_scan<<<1, 512, 0, stream>>>(xw, u0, b0, h1h, nullptr);

    // xw1 = h1 @ w1pad^T : [4096 x 512] * [512 x 512]^T
    k_wmma_gemm<<<dim3(HCAT / 64, T_DIM / 128), 256, 0, stream>>>(h1h, HCAT, w1p, HCAT,
                                                                  xw, HCAT, HCAT);
    k_indrnn_scan<<<1, 512, 0, stream>>>(xw, u1, b1, nullptr, h2f);

    k_mlp<<<dim3(T_DIM), 64, 0, stream>>>(h2f, mlp1w, mlp1b, mlp2w, mlp2b, pooled);
    k_mask<<<dim3(1), 1024, 0, stream>>>(pooled, maskp);

    // gx = relu(x) @ wih^T : [4096 x 4096] * [6144 x 4096]^T  (dominant GEMM)
    k_wmma_gemm<<<dim3(G3 / 64, T_DIM / 128), 256, 0, stream>>>(xh, F_DIM, wihh, F_DIM,
                                                                gx, G3, F_DIM);

    k_gru<<<dim3(GRU_NWG), 256, 0, stream>>>(gx, bih, whhh, bhh, maskp, hbuf, counter);
    k_final<<<dim3(1), 128, 0, stream>>>(hbuf + (T_DIM & 1) * GH, fc1w, fc1b, fc2w, fc2b, out);
}
